// GODIVA_5437428597352
// MI455X (gfx1250) — matrix operand order repaired and checked by reading, not compiled, not run on hardware
//
#include <hip/hip_runtime.h>
#include <cstdint>

typedef __attribute__((ext_vector_type(16))) __bf16 v16bf;
typedef __attribute__((ext_vector_type(8)))  float  v8f;
typedef __attribute__((ext_vector_type(2)))  __bf16 v2bf;

#define S_TOK 2048
#define NHEAD 16
#define DHEAD 64

// hardware bf16 conversion (v_cvt_pk_bf16_f32 path)
__device__ __forceinline__ uint32_t pack2(float a, float b) {
  union { v2bf h; uint32_t u; } x;
  x.h.x = (__bf16)a; x.h.y = (__bf16)b;
  return x.u;
}
__device__ __forceinline__ uint16_t f2bf(float f) {
  union { __bf16 h; uint16_t u; } x; x.h = (__bf16)f; return x.u;
}

union BF16Frag { v16bf v; uint32_t u[8]; };

// ---------------------------------------------------------------------------
// GEMM: C[2048,1024] = A[2048,1024] @ W[1024,1024] + bias
// MODE 0: out bf16, head-row layout  out[(h*2048 + s)*64 + d]   (Q, K)
// MODE 2: out bf16, head-transposed  out[(h*64 + d)*2048 + s]   (V)
// MODE 3: A is bf16 (attn), out f32  out[s*1024 + n]            (final proj)
// 256 thr (8 waves), tile 128(M) x 128(N), BK=32, double-buffered LDS.
// Waves 4x2; each wave owns 32x64 = 2x4 wmma 16x16 tiles.
// ---------------------------------------------------------------------------
template <int MODE>
__global__ __launch_bounds__(256)
void gemm_proj(const float* __restrict__ A_f32,
               const uint16_t* __restrict__ A_bf16,
               const float* __restrict__ W,
               const float* __restrict__ bias,
               void* __restrict__ out)
{
  __shared__ __align__(16) uint16_t Alds[2][128 * 40];  // row stride 80B (16B mult)
  __shared__ __align__(16) uint16_t Blds[2][128 * 40];

  const int tid  = threadIdx.x;
  const int lane = tid & 31;
  const int wid  = tid >> 5;
  const int wm   = wid & 3;          // 0..3 along M (32 rows each)
  const int wn   = wid >> 2;         // 0..1 along N (64 cols each)
  const int m0   = blockIdx.y * 128;
  const int n0   = blockIdx.x * 128;
  const int c    = lane & 15;
  const int hf   = lane >> 4;

  // staging ownership: A: row ar, 16 halves at col acs; B: col bc, 16 k at bk16
  const int ar   = tid >> 1;
  const int acs  = (tid & 1) * 16;
  const int bc   = tid & 127;
  const int bk16 = (tid >> 7) * 16;

  float    aregf[16];
  uint32_t aregb[8];
  float    bregf[16];

  auto fetch = [&](int k0) {
    if (MODE == 3) {
      const uint32_t* src =
          (const uint32_t*)(A_bf16 + (size_t)(m0 + ar) * 1024 + k0 + acs);
      #pragma unroll
      for (int q = 0; q < 8; ++q) aregb[q] = src[q];
    } else {
      const float* src = A_f32 + (size_t)(m0 + ar) * 1024 + k0 + acs;
      #pragma unroll
      for (int q = 0; q < 16; ++q) aregf[q] = src[q];
    }
    #pragma unroll
    for (int j = 0; j < 16; ++j)
      bregf[j] = W[(size_t)(k0 + bk16 + j) * 1024 + n0 + bc];
  };

  auto commit = [&](int buf) {
    uint32_t ap[8];
    if (MODE == 3) {
      #pragma unroll
      for (int q = 0; q < 8; ++q) ap[q] = aregb[q];
    } else {
      #pragma unroll
      for (int q = 0; q < 8; ++q) ap[q] = pack2(aregf[2 * q], aregf[2 * q + 1]);
    }
    uint4* arow = (uint4*)&Alds[buf][ar * 40 + acs];
    arow[0] = make_uint4(ap[0], ap[1], ap[2], ap[3]);
    arow[1] = make_uint4(ap[4], ap[5], ap[6], ap[7]);

    uint32_t bp[8];
    #pragma unroll
    for (int q = 0; q < 8; ++q) bp[q] = pack2(bregf[2 * q], bregf[2 * q + 1]);
    uint4* brow = (uint4*)&Blds[buf][bc * 40 + bk16];
    brow[0] = make_uint4(bp[0], bp[1], bp[2], bp[3]);
    brow[1] = make_uint4(bp[4], bp[5], bp[6], bp[7]);
  };

  v8f acc[2][4];
  #pragma unroll
  for (int i = 0; i < 2; ++i)
    #pragma unroll
    for (int j = 0; j < 4; ++j)
      #pragma unroll
      for (int e = 0; e < 8; ++e) acc[i][j][e] = 0.0f;

  fetch(0);
  for (int kt = 0; kt < 32; ++kt) {
    const int buf = kt & 1;
    commit(buf);
    __syncthreads();
    if (kt < 31) fetch((kt + 1) * 32);   // issue next tile's global loads early

    BF16Frag a[2], b[4];
    #pragma unroll
    for (int i = 0; i < 2; ++i) {
      const uint16_t* row = &Alds[buf][(wm * 32 + i * 16 + c) * 40];
      const int kb = hf * 16;            // A layout: K 0-15 / 16-31 by half
      #pragma unroll
      for (int q = 0; q < 4; ++q) {
        a[i].u[q]     = *(const uint32_t*)&row[kb + 2 * q];
        a[i].u[4 + q] = *(const uint32_t*)&row[kb + 8 + 2 * q];
      }
    }
    #pragma unroll
    for (int j = 0; j < 4; ++j) {
      const uint16_t* row = &Blds[buf][(wn * 64 + j * 16 + c) * 40];
      const int kb = hf * 16;            // B layout: K 0-15 / 16-31 by half
      #pragma unroll
      for (int q = 0; q < 8; ++q)
        b[j].u[q] = *(const uint32_t*)&row[kb + 2 * q];
    }
    #pragma unroll
    for (int i = 0; i < 2; ++i)
      #pragma unroll
      for (int j = 0; j < 4; ++j)
        acc[i][j] = __builtin_amdgcn_wmma_f32_16x16x32_bf16(
            false, a[i].v, false, b[j].v, (short)0, acc[i][j], false, false);
  }

  // ---- epilogue ----
  #pragma unroll
  for (int i = 0; i < 2; ++i) {
    #pragma unroll
    for (int j = 0; j < 4; ++j) {
      const int n  = n0 + wn * 64 + j * 16 + c;
      const float bv = bias[n];
      if (MODE == 2) {
        // d fixed per lane; 8 consecutive s -> one 16B store
        const int h = n >> 6, d = n & 63;
        const int s0 = m0 + wm * 32 + i * 16 + hf * 8;
        uint4 pk;
        pk.x = pack2(acc[i][j][0] + bv, acc[i][j][1] + bv);
        pk.y = pack2(acc[i][j][2] + bv, acc[i][j][3] + bv);
        pk.z = pack2(acc[i][j][4] + bv, acc[i][j][5] + bv);
        pk.w = pack2(acc[i][j][6] + bv, acc[i][j][7] + bv);
        *(uint4*)&((uint16_t*)out)[((size_t)h * 64 + d) * S_TOK + s0] = pk;
      } else {
        #pragma unroll
        for (int jr = 0; jr < 8; ++jr) {
          const int s = m0 + wm * 32 + i * 16 + jr + hf * 8;
          const float v = acc[i][j][jr] + bv;
          if (MODE == 3) {
            ((float*)out)[(size_t)s * 1024 + n] = v;
          } else {
            const int h = n >> 6, d = n & 63;
            ((uint16_t*)out)[((size_t)h * S_TOK + s) * 64 + d] = f2bf(v);
          }
        }
      }
    }
  }
}

// ---------------------------------------------------------------------------
// Sparse flash attention, one wave (32 thr) per (query-block, head).
// S^T = K @ Q^T (queries across lanes); softmax stats per-lane after one
// xor-16 reduction.  O^T = V^T @ P^T with K padded 16->32.  Mask computed
// analytically; only key blocks qb-k (k=0..15) and qb-16m (m=1..7) visited.
// ---------------------------------------------------------------------------
__global__ __launch_bounds__(32)
void attn_kernel(const uint16_t* __restrict__ qh,
                 const uint16_t* __restrict__ kh,
                 const uint16_t* __restrict__ vt,
                 uint16_t* __restrict__ attn_out)
{
  const int lane = threadIdx.x & 31;
  const int c  = lane & 15;
  const int hf = lane >> 4;
  const int qb = blockIdx.x;
  const int h  = blockIdx.y;
  const int q0 = qb * 16;

  // B-fragments of Q^T (N = query cols, K = d), kk = d-chunk 0..1
  BF16Frag bq[2];
  {
    const uint16_t* row = qh + ((size_t)h * S_TOK + q0 + c) * 64;
    #pragma unroll
    for (int kk = 0; kk < 2; ++kk) {
      const int kb = kk * 32 + hf * 16;
      #pragma unroll
      for (int q = 0; q < 8; ++q)
        bq[kk].u[q] = *(const uint32_t*)&row[kb + 2 * q];
    }
  }

  float mprev = -3.0e38f;
  float lsum  = 0.0f;
  v8f o[4];
  #pragma unroll
  for (int dd = 0; dd < 4; ++dd)
    #pragma unroll
    for (int e = 0; e < 8; ++e) o[dd][e] = 0.0f;

  const int i_tok = q0 + c;
  const int ti = i_tok >> 8, hi = (i_tok >> 4) & 15, wi = i_tok & 15;

  auto process = [&](int kblk) {
    const int t0 = kblk * 16;

    // A-fragments of K (rows = keys, K-dim = d)
    BF16Frag ak[2];
    const uint16_t* krow = kh + ((size_t)h * S_TOK + t0 + c) * 64;
    #pragma unroll
    for (int kk = 0; kk < 2; ++kk) {
      const int kb = kk * 32 + hf * 16;
      #pragma unroll
      for (int q = 0; q < 4; ++q) {
        ak[kk].u[q]     = *(const uint32_t*)&krow[kb + 2 * q];
        ak[kk].u[4 + q] = *(const uint32_t*)&krow[kb + 8 + 2 * q];
      }
    }
    v8f st;
    #pragma unroll
    for (int e = 0; e < 8; ++e) st[e] = 0.0f;
    st = __builtin_amdgcn_wmma_f32_16x16x32_bf16(false, ak[0].v, false, bq[0].v,
                                                 (short)0, st, false, false);
    st = __builtin_amdgcn_wmma_f32_16x16x32_bf16(false, ak[1].v, false, bq[1].v,
                                                 (short)0, st, false, false);

    // issue all V^T fragment loads now: softmax VALU below hides their latency
    BF16Frag av[4];
    #pragma unroll
    for (int dd = 0; dd < 4; ++dd) {
      const uint16_t* vrow = vt + ((size_t)h * 64 + dd * 16 + c) * S_TOK + t0;
      #pragma unroll
      for (int q = 0; q < 4; ++q) {
        av[dd].u[q]     = *(const uint32_t*)&vrow[hf * 8 + 2 * q];
        av[dd].u[4 + q] = 0u;          // K=16..31 zero pad
      }
    }

    // analytic mask + scale; stats per query column (= per lane)
    float p[8];
    float rmax = -3.0e38f;
    #pragma unroll
    for (int jr = 0; jr < 8; ++jr) {
      const int j_tok = t0 + jr + hf * 8;
      const int diff  = i_tok - j_tok;
      const int tj = j_tok >> 8, hj = (j_tok >> 4) & 15, wj = j_tok & 15;
      const bool m = (diff >= 0 && diff <= 3)
                  || (hi == hj && wi == wj && tj <= ti)
                  || (ti == tj && wi == wj && hj <= hi);
      const float sv = m ? st[jr] * 0.125f : -1.0e9f;
      p[jr] = sv;
      rmax  = fmaxf(rmax, sv);
    }
    rmax = fmaxf(rmax, __shfl_xor(rmax, 16));
    const float mnew  = fmaxf(mprev, rmax);
    const float alpha = __expf(mprev - mnew);
    float rsum = 0.0f;
    #pragma unroll
    for (int jr = 0; jr < 8; ++jr) { p[jr] = __expf(p[jr] - mnew); rsum += p[jr]; }
    rsum += __shfl_xor(rsum, 16);
    lsum  = lsum * alpha + rsum;
    mprev = mnew;
    #pragma unroll
    for (int dd = 0; dd < 4; ++dd)
      #pragma unroll
      for (int jr = 0; jr < 8; ++jr) o[dd][jr] *= alpha;

    // B-fragment of P^T: lanes 0-15 carry K(t)=0..7, lanes 16-31 K(t)=8..15, elements 8-15 zero pad
    float po[8];
    #pragma unroll
    for (int jr = 0; jr < 8; ++jr) po[jr] = __shfl_xor(p[jr], 16);
    BF16Frag bp;
    #pragma unroll
    for (int q = 0; q < 4; ++q) {
      bp.u[q]     = pack2(p[2 * q],  p[2 * q + 1]);
      bp.u[4 + q] = 0u;
    }

    // O^T += V^T @ P^T  (4 d-tiles of 16)
    #pragma unroll
    for (int dd = 0; dd < 4; ++dd)
      o[dd] = __builtin_amdgcn_wmma_f32_16x16x32_bf16(
          false, av[dd].v, false, bp.v, (short)0, o[dd], false, false);
  };

  for (int k = 0; k <= 15; ++k) { const int kb = qb - k;      if (kb < 0) break; process(kb); }
  for (int m = 1; m <= 7;  ++m) { const int kb = qb - 16 * m; if (kb < 0) break; process(kb); }

  // normalize and store O^T -> attn_out[s][h*64+d] (bf16, 16B stores)
  const float inv = 1.0f / lsum;
  #pragma unroll
  for (int dd = 0; dd < 4; ++dd) {
    uint4 pk;
    pk.x = pack2(o[dd][0] * inv, o[dd][1] * inv);
    pk.y = pack2(o[dd][2] * inv, o[dd][3] * inv);
    pk.z = pack2(o[dd][4] * inv, o[dd][5] * inv);
    pk.w = pack2(o[dd][6] * inv, o[dd][7] * inv);
    uint16_t* dst = attn_out + (size_t)(q0 + c) * 1024 + h * 64 + dd * 16 + hf * 8;
    *(uint4*)dst = pk;
  }
}

// ---------------------------------------------------------------------------
extern "C" void kernel_launch(void* const* d_in, const int* in_sizes, int n_in,
                              void* d_out, int out_size, void* d_ws, size_t ws_size,
                              hipStream_t stream) {
  const float* query = (const float*)d_in[0];
  const float* key   = (const float*)d_in[1];
  const float* value = (const float*)d_in[2];
  const float* wq = (const float*)d_in[3];
  const float* bqv = (const float*)d_in[4];
  const float* wk = (const float*)d_in[5];
  const float* bkv = (const float*)d_in[6];
  const float* wv = (const float*)d_in[7];
  const float* bvv = (const float*)d_in[8];
  const float* wo = (const float*)d_in[9];
  const float* bo = (const float*)d_in[10];
  // d_in[11] = layout: intentionally unused (mask reconstructed analytically)

  const size_t HRE = (size_t)NHEAD * S_TOK * DHEAD;   // elems per bf16 region
  uint16_t* qh   = (uint16_t*)d_ws;
  uint16_t* kh   = qh + HRE;
  uint16_t* vt   = kh + HRE;
  uint16_t* attn = vt + HRE;                          // [2048][1024] bf16

  dim3 gg(8, 16), gb(256);
  gemm_proj<0><<<gg, gb, 0, stream>>>(query, nullptr, wq, bqv, qh);
  gemm_proj<0><<<gg, gb, 0, stream>>>(key,   nullptr, wk, bkv, kh);
  gemm_proj<2><<<gg, gb, 0, stream>>>(value, nullptr, wv, bvv, vt);

  attn_kernel<<<dim3(S_TOK / 16, NHEAD), 32, 0, stream>>>(qh, kh, vt, attn);

  gemm_proj<3><<<gg, gb, 0, stream>>>(nullptr, attn, wo, bo, d_out);
}
